// GATLayer_80814104642373
// MI455X (gfx1250) — matrix operand order repaired and checked by reading, not compiled, not run on hardware
//
#include <hip/hip_runtime.h>
#include <stdint.h>

// Problem constants (match reference)
constexpr int N  = 50000;     // nodes (= 3125 * 16, exact M-tiles)
constexpr int E  = 800000;    // edges
constexpr int H  = 4;         // heads
constexpr int D  = 32;        // per-head dim
constexpr int IN = 128;       // features
constexpr int N128 = N * IN;  // 6,400,000
constexpr int NH   = N * H;   // 200,000
constexpr int EH   = E * H;   // 3,200,000

typedef __attribute__((ext_vector_type(16))) _Float16 v16h;
typedef __attribute__((ext_vector_type(8)))  float    v8f;
typedef __attribute__((ext_vector_type(4)))  int      v4i;

#define HAVE_ASYNC_LDS (__has_builtin(__builtin_amdgcn_global_load_async_to_lds_b128) && \
                        __has_builtin(__builtin_amdgcn_s_wait_asynccnt))

// ---------------------------------------------------------------------------
// WMMA fragment loaders per CDNA5 ISA 7.12.2 (wave32 layouts)
// A: 16x32 f16 (MxK). lane m = l&15 ; K-base = (l&16)?8:0
//    dword i   (i<4) -> K = kb + 2i,2i+1
//    dword 4+i       -> K = kb + 16 + 2i, +1
// ---------------------------------------------------------------------------
__device__ __forceinline__ v16h load_a16x32(const _Float16* tile, int ld, int k0, int lane) {
    int m  = lane & 15;
    int kb = (lane & 16) ? 8 : 0;
    const uint32_t* p = (const uint32_t*)(tile + m * ld + k0 + kb);
    union { uint32_t u[8]; v16h h; } f;
#pragma unroll
    for (int i = 0; i < 4; ++i) { f.u[i] = p[i]; f.u[4 + i] = p[8 + i]; }
    return f.h;
}

// B: 32x16 f16 (KxN), B[k][n] = W[n][k] with W row-major (out x K).
// lane col n = ocol + (l&15) ; K-base = (l&16)?8:0 ; dword i<4 -> K = kb+2i,+1 ; dword 4+i -> K = kb+16+2i,+1
__device__ __forceinline__ v16h load_b32x16(const _Float16* W, int ld, int ocol, int k0, int lane) {
    int o  = ocol + (lane & 15);
    int kb = (lane & 16) ? 8 : 0;
    const uint32_t* p = (const uint32_t*)(W + o * ld + k0 + kb);
    union { uint32_t u[8]; v16h h; } f;
#pragma unroll
    for (int i = 0; i < 4; ++i) { f.u[i] = p[i]; f.u[4 + i] = p[8 + i]; }
    return f.h;
}

// ---------------------------------------------------------------------------
// K0: convert Wq/Wk/Wv to f16; head-reduce Wscale -> Wsum (128x128) f16
// w_h layout: [0]=Wq [1]=Wk [2]=Wv [3]=Wsum, each 128x128 row-major (out x K)
// ---------------------------------------------------------------------------
__global__ void k_convert_w(const float* __restrict__ Wq, const float* __restrict__ Wk,
                            const float* __restrict__ Wv, const float* __restrict__ Wscale,
                            _Float16* __restrict__ w_h) {
    int i = blockIdx.x * blockDim.x + threadIdx.x;      // 0 .. 128*128-1
    w_h[0 * 16384 + i] = (_Float16)Wq[i];
    w_h[1 * 16384 + i] = (_Float16)Wk[i];
    w_h[2 * 16384 + i] = (_Float16)Wv[i];
    int j = i >> 7, kk = i & 127;                       // Wsum[j][k] = sum_h Wscale[h*128+j][k]
    float s = 0.f;
#pragma unroll
    for (int hh = 0; hh < H; ++hh) s += Wscale[((size_t)(hh * 128 + j)) * 128 + kk];
    w_h[3 * 16384 + i] = (_Float16)s;
}

__global__ void k_convert_feat(const float* __restrict__ f, _Float16* __restrict__ fh) {
    int i = blockIdx.x * blockDim.x + threadIdx.x;
    fh[i] = (_Float16)f[i];
}

__global__ void k_zero(float* __restrict__ ft, float* __restrict__ den,
                       unsigned* __restrict__ nmax) {
    int t = blockIdx.x * blockDim.x + threadIdx.x;
    if (t < N128)            ft[t] = 0.f;
    else if (t < N128 + NH)  den[t - N128] = 0.f;
    else if (t < N128 + 2 * NH) nmax[t - N128 - NH] = 0u;   // below enc(-inf): safe floor
}

// ---------------------------------------------------------------------------
// K1: fused q/k/v = feat @ W{q,k,v}^T via v_wmma_f32_16x16x32_f16.
// grid = N/16 blocks, 256 thr = 8 waves; wave w owns cols [16w,16w+16) of all
// THREE output matrices, reusing one A fragment for 3 WMMAs per K-step.
// A tile (16x128 f16 = 4KB) staged in LDS via ASYNC-to-LDS when available.
// ---------------------------------------------------------------------------
__global__ void k_gemm_qkv(const _Float16* __restrict__ feat_h,
                           const _Float16* __restrict__ w_h,
                           float* __restrict__ qkv) {
    __shared__ _Float16 smA[16 * 128];
    const int tid = threadIdx.x, lane = tid & 31, wave = tid >> 5;
    const size_t row0 = (size_t)blockIdx.x * 16;

#if HAVE_ASYNC_LDS
    {   // 256 lanes x 16B async copies, tracked by ASYNCcnt
        __attribute__((address_space(1))) v4i* g =
            (__attribute__((address_space(1))) v4i*)(feat_h + row0 * 128 + (size_t)tid * 8);
        __attribute__((address_space(3))) v4i* l =
            (__attribute__((address_space(3))) v4i*)(&smA[tid * 8]);
        __builtin_amdgcn_global_load_async_to_lds_b128(g, l, 0, 0);
        __builtin_amdgcn_s_wait_asynccnt(0);
    }
#else
    ((uint4*)smA)[tid] = ((const uint4*)(feat_h + row0 * 128))[tid];
#endif
    __syncthreads();

    const int ocol = wave * 16;
    v8f c0 = {0.f, 0.f, 0.f, 0.f, 0.f, 0.f, 0.f, 0.f};
    v8f c1 = c0, c2 = c0;
#pragma unroll
    for (int k0 = 0; k0 < 128; k0 += 32) {
        v16h a  = load_a16x32(smA, 128, k0, lane);
        v16h b0 = load_b32x16(w_h + 0 * 16384, 128, ocol, k0, lane);
        v16h b1 = load_b32x16(w_h + 1 * 16384, 128, ocol, k0, lane);
        v16h b2 = load_b32x16(w_h + 2 * 16384, 128, ocol, k0, lane);
        c0 = __builtin_amdgcn_wmma_f32_16x16x32_f16(false, a, false, b0, (short)0, c0, false, false);
        c1 = __builtin_amdgcn_wmma_f32_16x16x32_f16(false, a, false, b1, (short)0, c1, false, false);
        c2 = __builtin_amdgcn_wmma_f32_16x16x32_f16(false, a, false, b2, (short)0, c2, false, false);
    }
    const int n = ocol + (lane & 15);
    const int mbase = (lane & 16) ? 8 : 0;
    float* q = qkv;
    float* k = qkv + (size_t)N128;
    float* v = qkv + (size_t)2 * N128;
#pragma unroll
    for (int r = 0; r < 8; ++r) {
        size_t idx = (row0 + mbase + r) * 128 + n;
        q[idx] = c0[r];
        k[idx] = c1[r];
        v[idx] = c2[r];
    }
}

// ---------------------------------------------------------------------------
// K2: per (edge, head) score = dot32(q[src],k[dst]) / sqrt(D); segment max via
// order-preserving uint atomicMax.
// ---------------------------------------------------------------------------
__device__ __forceinline__ unsigned enc_f32(float f) {
    unsigned u = __float_as_uint(f);
    return (u & 0x80000000u) ? ~u : (u | 0x80000000u);
}
__device__ __forceinline__ float dec_f32(unsigned u) {
    return __uint_as_float((u & 0x80000000u) ? (u ^ 0x80000000u) : ~u);
}

__global__ void k_scores(const float* __restrict__ q, const float* __restrict__ k,
                         const int* __restrict__ src, const int* __restrict__ dst,
                         float* __restrict__ sc, unsigned* __restrict__ nmax) {
    int t = blockIdx.x * blockDim.x + threadIdx.x;      // EH threads exactly
    int e = t >> 2, h = t & 3;
    int s = src[e], d = dst[e];
    const float4* qp = (const float4*)(q + (size_t)s * 128 + h * 32);
    const float4* kp = (const float4*)(k + (size_t)d * 128 + h * 32);
    float acc = 0.f;
#pragma unroll
    for (int i = 0; i < 8; ++i) {
        float4 a = qp[i], b = kp[i];
        acc += a.x * b.x + a.y * b.y + a.z * b.z + a.w * b.w;
    }
    acc *= 0.17677669529663687f;                        // 32^-0.5
    sc[t] = acc;
    atomicMax(&nmax[(size_t)d * H + h], enc_f32(acc));
}

// K3: ex = exp(s - max[dst]); segment sum
__global__ void k_expsum(float* __restrict__ sc, const int* __restrict__ dst,
                         const unsigned* __restrict__ nmax, float* __restrict__ den) {
    int t = blockIdx.x * blockDim.x + threadIdx.x;
    int e = t >> 2, h = t & 3;
    int d = dst[e];
    float ex = expf(sc[t] - dec_f32(nmax[(size_t)d * H + h]));
    sc[t] = ex;                                         // overwrite in place
    atomicAdd(&den[(size_t)d * H + h], ex);
}

// K4: a = ex/den[dst]; scatter v[src]*a into ft[dst]
__global__ void k_aggregate(const float* __restrict__ sc, const float* __restrict__ den,
                            const float* __restrict__ v, const int* __restrict__ src,
                            const int* __restrict__ dst, float* __restrict__ ft) {
    int t = blockIdx.x * blockDim.x + threadIdx.x;
    int e = t >> 2, h = t & 3;
    int s = src[e], d = dst[e];
    float a = sc[t] / den[(size_t)d * H + h];
    const float* vp = v + (size_t)s * 128 + h * 32;
    float* fp = ft + (size_t)d * 128 + h * 32;
#pragma unroll
    for (int i = 0; i < 32; ++i) atomicAdd(&fp[i], vp[i] * a);
}

// ---------------------------------------------------------------------------
// K5: out = ft @ Wsum^T + feature  (f32 ft packed to f16 in LDS on the fly)
// ---------------------------------------------------------------------------
__global__ void k_gemm_out(const float* __restrict__ ft, const _Float16* __restrict__ Wsum,
                           const float* __restrict__ feature, float* __restrict__ out) {
    __shared__ _Float16 smA[16 * 128];
    const int tid = threadIdx.x, lane = tid & 31, wave = tid >> 5;
    const size_t row0 = (size_t)blockIdx.x * 16;

    // stage + convert: 2048 f32 -> 1024 packed dwords, 4 per thread
    {
        const float2* sp = (const float2*)(ft + row0 * 128);
        uint32_t* dp = (uint32_t*)smA;
#pragma unroll
        for (int i = 0; i < 4; ++i) {
            float2 f = sp[tid + i * 256];
            union { uint32_t u; _Float16 h[2]; } pk;
            pk.h[0] = (_Float16)f.x;
            pk.h[1] = (_Float16)f.y;
            dp[tid + i * 256] = pk.u;
        }
    }
    __syncthreads();

    const int ocol = wave * 16;
    v8f c = {0.f, 0.f, 0.f, 0.f, 0.f, 0.f, 0.f, 0.f};
#pragma unroll
    for (int k0 = 0; k0 < 128; k0 += 32) {
        v16h a = load_a16x32(smA, 128, k0, lane);
        v16h b = load_b32x16(Wsum, 128, ocol, k0, lane);
        c = __builtin_amdgcn_wmma_f32_16x16x32_f16(false, a, false, b, (short)0, c, false, false);
    }
    const int n = ocol + (lane & 15);
    const int mbase = (lane & 16) ? 8 : 0;
#pragma unroll
    for (int r = 0; r < 8; ++r) {
        size_t idx = (row0 + mbase + r) * 128 + n;
        out[idx] = c[r] + feature[idx];
    }
}

// ---------------------------------------------------------------------------
extern "C" void kernel_launch(void* const* d_in, const int* in_sizes, int n_in,
                              void* d_out, int out_size, void* d_ws, size_t ws_size,
                              hipStream_t stream) {
    (void)in_sizes; (void)n_in; (void)out_size; (void)ws_size;
    const float* feature = (const float*)d_in[0];
    const float* Wq      = (const float*)d_in[1];
    const float* Wk      = (const float*)d_in[2];
    const float* Wv      = (const float*)d_in[3];
    const float* Wscale  = (const float*)d_in[4];
    const int*   src     = (const int*)d_in[5];
    const int*   dst     = (const int*)d_in[6];
    float* out = (float*)d_out;

    char* ws = (char*)d_ws;
    size_t off = 0;
    auto take = [&](size_t bytes) -> char* {
        char* p = ws + off;
        off += (bytes + 255) & ~(size_t)255;
        return p;
    };
    _Float16* feat_h = (_Float16*)take((size_t)N128 * 2);          // 12.8 MB
    _Float16* w_h    = (_Float16*)take((size_t)4 * 16384 * 2);     // 128 KB
    float*    qkv    = (float*)   take((size_t)3 * N128 * 4);      // 76.8 MB
    float*    ft     = (float*)   take((size_t)N128 * 4);          // 25.6 MB
    float*    sc     = (float*)   take((size_t)EH * 4);            // 12.8 MB
    unsigned* nmax   = (unsigned*)take((size_t)NH * 4);            // 0.8 MB
    float*    den    = (float*)   take((size_t)NH * 4);            // 0.8 MB

    k_convert_w   <<<16384 / 256, 256, 0, stream>>>(Wq, Wk, Wv, Wscale, w_h);
    k_convert_feat<<<N128 / 256, 256, 0, stream>>>(feature, feat_h);
    k_zero        <<<(N128 + 2 * NH + 255) / 256, 256, 0, stream>>>(ft, den, nmax);
    k_gemm_qkv    <<<N / 16, 256, 0, stream>>>(feat_h, w_h, qkv);
    k_scores      <<<EH / 256, 256, 0, stream>>>(qkv, qkv + (size_t)N128, src, dst, sc, nmax);
    k_expsum      <<<EH / 256, 256, 0, stream>>>(sc, dst, nmax, den);
    k_aggregate   <<<EH / 256, 256, 0, stream>>>(sc, den, qkv + (size_t)2 * N128, src, dst, ft);
    k_gemm_out    <<<N / 16, 256, 0, stream>>>(ft, w_h + 3 * 16384, feature, out);
}
